// ShiftedChannelMumtiHead_78288663872049
// MI455X (gfx1250) — matrix operand order repaired and checked by reading, not compiled, not run on hardware
//
#include <hip/hip_runtime.h>
#include <hip/hip_bf16.h>
#include <math.h>

#define BB 4
#define CC 256
#define NN 2048
#define WW 64
#define DD 32
#define HH 7
#define FF (HH*WW)   /* 448 */
#define SROW 2056    /* 2048 + 8 bf16 pad to dodge LDS bank aliasing */

typedef __attribute__((ext_vector_type(16))) __bf16 v16bf;
typedef __attribute__((ext_vector_type(8)))  float  v8f;

__device__ __forceinline__ __bf16 f2bf(float f){ return (__bf16)f; }
__device__ __forceinline__ float  bf2f(__bf16 b){ return (float)b; }

// A fragment (16x32 bf16). rowptr = &A[M][k0] for this lane's row (M = lane&15).
// Lane half kh: elements 0..15 <- K = 16*kh..16*kh+15
__device__ __forceinline__ v16bf load_a(const __bf16* rowptr, int kh){
  v16bf a;
  const __bf16* p0 = rowptr + 16*kh;
  const __bf16* p1 = rowptr + 16*kh + 8;
  #pragma unroll
  for (int i=0;i<8;i++){ a[i]=p0[i]; a[8+i]=p1[i]; }
  return a;
}
// B fragment (32x16). p points at the 16 K-contiguous values for this lane's
// column (N = lane&15), already offset by 16*kh.
__device__ __forceinline__ v16bf load_b(const __bf16* p){
  v16bf b;
  #pragma unroll
  for (int i=0;i<16;i++) b[i]=p[i];
  return b;
}

#define WMMA(a,b,c) __builtin_amdgcn_wmma_f32_16x16x32_bf16(false,(a),false,(b),(short)0,(c),false,false)

// ---------------------------------------------------------------------------
// Kernel 0: elementwise f32 -> bf16 weight conversion (one-time, tiny)
// ---------------------------------------------------------------------------
__global__ void k_cvt(const float* __restrict__ in, __bf16* __restrict__ out, int n){
  int i = blockIdx.x*256 + threadIdx.x;
  if (i < n) out[i] = f2bf(in[i]);
}

// ---------------------------------------------------------------------------
// Kernel 1: xT[b][n][c] = bf16(x[b][c][n])  (LDS tile transpose; makes every
// later fragment load contiguous)
// ---------------------------------------------------------------------------
__global__ void k_transpose(const float* __restrict__ x, __bf16* __restrict__ xT){
  __shared__ float tile[32][33];
  int b = blockIdx.z, c0 = blockIdx.y*32, n0 = blockIdx.x*32;
  int tx = threadIdx.x, ty = threadIdx.y; // 32 x 8
  #pragma unroll
  for(int i=0;i<4;i++)
    tile[ty+8*i][tx] = x[((size_t)b*CC + c0+ty+8*i)*NN + n0+tx];
  __syncthreads();
  #pragma unroll
  for(int i=0;i<4;i++)
    xT[((size_t)b*NN + n0+ty+8*i)*CC + c0+tx] = f2bf(tile[tx][ty+8*i]);
}

// ---------------------------------------------------------------------------
// Kernel 2: q/k/v conv_bn_relu GEMMs, all-bf16 operands.  Each wave computes a
// 2x2 grid of 16x16 D tiles (4 WMMA per k-step from 2 A + 2 B fragments).
// mode 0/1 -> transposed outputs qT/kT (B,N,C); mode 2 -> v natural (B,C,N).
// Block = 256 threads = 8 waves; WG tile 128 x 64; K = 256 = 8 k-steps.
// ---------------------------------------------------------------------------
__global__ void k_qkv(const __bf16* __restrict__ xT,
                      const __bf16* __restrict__ Wqb, const __bf16* __restrict__ Wkb,
                      const __bf16* __restrict__ Wvb,
                      const float* __restrict__ sq, const float* __restrict__ bq,
                      const float* __restrict__ sk, const float* __restrict__ bk,
                      const float* __restrict__ sv, const float* __restrict__ bv,
                      __bf16* __restrict__ qT, __bf16* __restrict__ kT,
                      __bf16* __restrict__ vN)
{
  int mode = blockIdx.z % 3, b = blockIdx.z / 3;
  int lane = threadIdx.x & 31, wid = threadIdx.x >> 5;
  int kh = lane >> 4, lx = lane & 15;
  const __bf16* Wm = mode==0?Wqb: mode==1?Wkb:Wvb;
  const float*  sm = mode==0?sq : mode==1?sk :sv;
  const float*  bm = mode==0?bq : mode==1?bk :bv;

  v8f acc[2][2] = {};
  if (mode < 2){
    int n0 = blockIdx.x*128 + (wid & 3)*32;
    int o0 = blockIdx.y*64  + (wid >> 2)*32;
    const __bf16* ar0 = xT + ((size_t)b*NN + n0      + lx)*CC;
    const __bf16* ar1 = xT + ((size_t)b*NN + n0 + 16 + lx)*CC;
    const __bf16* bc0 = Wm + (size_t)(o0      + lx)*CC + 16*kh;
    const __bf16* bc1 = Wm + (size_t)(o0 + 16 + lx)*CC + 16*kh;
    #pragma unroll 2
    for(int s=0;s<8;s++){
      __builtin_prefetch(ar0 + (s+1)*32, 0, 1);
      v16bf a0 = load_a(ar0 + s*32, kh);
      v16bf a1 = load_a(ar1 + s*32, kh);
      v16bf b0 = load_b(bc0 + s*32);
      v16bf b1 = load_b(bc1 + s*32);
      acc[0][0] = WMMA(a0, b0, acc[0][0]);
      acc[0][1] = WMMA(a0, b1, acc[0][1]);
      acc[1][0] = WMMA(a1, b0, acc[1][0]);
      acc[1][1] = WMMA(a1, b1, acc[1][1]);
    }
    __bf16* outT = (mode==0) ? qT : kT;
    #pragma unroll
    for(int j=0;j<2;j++){
      int oc = o0 + j*16 + lx;
      float sc = sm[oc], bi = bm[oc];
      #pragma unroll
      for(int i=0;i<2;i++){
        #pragma unroll
        for(int r=0;r<8;r++){
          int n = n0 + i*16 + r + 8*kh;
          outT[((size_t)b*NN + n)*CC + oc] = f2bf(fmaxf(acc[i][j][r]*sc + bi, 0.f));
        }
      }
    }
  } else {
    int m0 = blockIdx.y*64  + (wid >> 2)*32;   // output-channel rows
    int n0 = blockIdx.x*128 + (wid & 3)*32;    // output columns
    const __bf16* ar0 = Wm + (size_t)(m0      + lx)*CC;
    const __bf16* ar1 = Wm + (size_t)(m0 + 16 + lx)*CC;
    const __bf16* bc0 = xT + ((size_t)b*NN + n0      + lx)*CC + 16*kh;
    const __bf16* bc1 = xT + ((size_t)b*NN + n0 + 16 + lx)*CC + 16*kh;
    #pragma unroll 2
    for(int s=0;s<8;s++){
      __builtin_prefetch(bc0 + (s+1)*32, 0, 1);
      v16bf a0 = load_a(ar0 + s*32, kh);
      v16bf a1 = load_a(ar1 + s*32, kh);
      v16bf b0 = load_b(bc0 + s*32);
      v16bf b1 = load_b(bc1 + s*32);
      acc[0][0] = WMMA(a0, b0, acc[0][0]);
      acc[0][1] = WMMA(a0, b1, acc[0][1]);
      acc[1][0] = WMMA(a1, b0, acc[1][0]);
      acc[1][1] = WMMA(a1, b1, acc[1][1]);
    }
    #pragma unroll
    for(int i=0;i<2;i++){
      #pragma unroll
      for(int r=0;r<8;r++){
        int ch = m0 + i*16 + r + 8*kh;
        float sc = sm[ch], bi = bm[ch];
        #pragma unroll
        for(int j=0;j<2;j++){
          int n = n0 + j*16 + lx;
          vN[((size_t)b*CC + ch)*NN + n] = f2bf(fmaxf(acc[i][j][r]*sc + bi, 0.f));
        }
      }
    }
  }
}

// ---------------------------------------------------------------------------
// Kernel 3: attention for one (b, head, 32-row n-strip).  Score strip S
// (32 x 2048 bf16) lives entirely in LDS (CDNA5: 320KB/WGP), softmax is
// exact, then O^T = P * V^T accumulates via WMMA and writes straight into
// OT (B,N,448) bf16 — exactly the B-operand layout the final GEMM wants.
// ---------------------------------------------------------------------------
__global__ void k_attn(const __bf16* __restrict__ qT,
                       const __bf16* __restrict__ kT,
                       const __bf16* __restrict__ vN,
                       __bf16* __restrict__ OT)
{
  __shared__ __bf16 S[32*SROW];   // 131584 B
  __shared__ float red[32][8];
  __shared__ float rowstat[32];

  int b = blockIdx.z, head = blockIdx.y;
  int n0 = blockIdx.x*32;
  int tid  = threadIdx.x;
  int lane = tid & 31, wid = tid >> 5;
  int kh = lane >> 4, lx = lane & 15;
  int cbase = head*DD;                    // window channel base

  // ---- Stage 1: S = Qw^T * Kw ; waves partition the 2048 m-columns ----
  v16bf afr[2][2];
  #pragma unroll
  for(int ns=0; ns<2; ns++){
    const __bf16* arow = qT + ((size_t)b*NN + n0 + ns*16 + lx)*CC + cbase;
    afr[ns][0] = load_a(arow,      kh);
    afr[ns][1] = load_a(arow + 32, kh);
  }
  #pragma unroll 1
  for(int mt=0; mt<16; mt++){
    int m0 = wid*256 + mt*16;
    const __bf16* bcol = kT + ((size_t)b*NN + m0 + lx)*CC + cbase + 16*kh;
    __builtin_prefetch(bcol + 16*CC, 0, 1);
    v16bf b0 = load_b(bcol);
    v16bf b1 = load_b(bcol + 32);
    #pragma unroll
    for(int ns=0; ns<2; ns++){
      v8f acc = {};
      acc = WMMA(afr[ns][0], b0, acc);
      acc = WMMA(afr[ns][1], b1, acc);
      #pragma unroll
      for(int r=0;r<8;r++){
        int row = ns*16 + r + 8*kh;
        S[row*SROW + m0 + lx] = f2bf(acc[r]);
      }
    }
  }
  __syncthreads();

  // ---- Stage 2: exact softmax stats; P = exp(S - rowmax) kept unnormalized,
  //      rowsum folded into stage-3 epilogue ----
  int row = tid & 31, seg = tid >> 5, c0 = seg*256;
  float mx = -3.0e38f;
  for(int j=0;j<256;j++) mx = fmaxf(mx, bf2f(S[row*SROW + c0 + j]));
  red[row][seg] = mx;
  __syncthreads();
  if (tid < 32){
    float m2 = red[tid][0];
    #pragma unroll
    for(int j=1;j<8;j++) m2 = fmaxf(m2, red[tid][j]);
    rowstat[tid] = m2;
  }
  __syncthreads();
  float rm = rowstat[row];
  float sum = 0.f;
  for(int j=0;j<256;j++){
    float p = __expf(bf2f(S[row*SROW + c0 + j]) - rm);
    S[row*SROW + c0 + j] = f2bf(p);
    sum += p;
  }
  __syncthreads();
  red[row][seg] = sum;
  __syncthreads();
  if (tid < 32){
    float s2 = 0.f;
    #pragma unroll
    for(int j=0;j<8;j++) s2 += red[tid][j];
    rowstat[tid] = s2;       // rowsum
  }
  __syncthreads();

  // ---- Stage 3: O^T(32n x 64w) = P(32 x 2048) * V^T ; 8 waves = 2 nsub x 4 wsub
  int nsub = wid & 1, wsub = wid >> 1;
  int chan = cbase + wsub*16 + lx;
  const __bf16* bvrow = vN + ((size_t)b*CC + chan)*NN + 16*kh;
  const __bf16* prow  = S + (nsub*16 + lx)*SROW;
  v8f acc = {};
  #pragma unroll 2
  for(int ks=0; ks<64; ks++){
    __builtin_prefetch(bvrow + (ks+4)*32, 0, 1);
    v16bf a  = load_a(prow  + ks*32, kh);
    v16bf bb = load_b(bvrow + ks*32);
    acc = WMMA(a, bb, acc);
  }
  #pragma unroll
  for(int r=0;r<8;r++){
    int nl = nsub*16 + r + 8*kh;
    float vv = acc[r] / rowstat[nl];
    OT[((size_t)b*NN + n0 + nl)*FF + head*WW + wsub*16 + lx] = f2bf(vv);
  }
}

// ---------------------------------------------------------------------------
// Kernel 4: y = relu((Wf @ O + bf)*sf + bff).  All-bf16 operands; each wave
// computes a 2x2 tile grid; K = 448 = 14 k-steps.
// ---------------------------------------------------------------------------
__global__ void k_final(const __bf16* __restrict__ OT,
                        const __bf16* __restrict__ Wfb,
                        const float* __restrict__ bfv,
                        const float* __restrict__ sf, const float* __restrict__ bff,
                        float* __restrict__ out)
{
  int b = blockIdx.z;
  int lane = threadIdx.x & 31, wid = threadIdx.x >> 5;
  int kh = lane >> 4, lx = lane & 15;
  int n0 = blockIdx.x*128 + (wid & 3)*32;
  int o0 = blockIdx.y*64  + (wid >> 2)*32;
  const __bf16* ar0 = Wfb + (size_t)(o0      + lx)*FF;
  const __bf16* ar1 = Wfb + (size_t)(o0 + 16 + lx)*FF;
  const __bf16* bc0 = OT + ((size_t)b*NN + n0      + lx)*FF + 16*kh;
  const __bf16* bc1 = OT + ((size_t)b*NN + n0 + 16 + lx)*FF + 16*kh;
  v8f acc[2][2] = {};
  #pragma unroll 2
  for(int s=0;s<14;s++){
    __builtin_prefetch(bc0 + (s+1)*32, 0, 1);
    v16bf a0 = load_a(ar0 + s*32, kh);
    v16bf a1 = load_a(ar1 + s*32, kh);
    v16bf b0 = load_b(bc0 + s*32);
    v16bf b1 = load_b(bc1 + s*32);
    acc[0][0] = WMMA(a0, b0, acc[0][0]);
    acc[0][1] = WMMA(a0, b1, acc[0][1]);
    acc[1][0] = WMMA(a1, b0, acc[1][0]);
    acc[1][1] = WMMA(a1, b1, acc[1][1]);
  }
  #pragma unroll
  for(int i=0;i<2;i++){
    #pragma unroll
    for(int r=0;r<8;r++){
      int o = o0 + i*16 + r + 8*kh;
      float bi = bfv[o], sc = sf[o], b2 = bff[o];
      #pragma unroll
      for(int j=0;j<2;j++){
        int n = n0 + j*16 + lx;
        float y = acc[i][j][r] + bi;
        out[((size_t)b*CC + o)*NN + n] = fmaxf(y*sc + b2, 0.f);
      }
    }
  }
}

// ---------------------------------------------------------------------------
extern "C" void kernel_launch(void* const* d_in, const int* in_sizes, int n_in,
                              void* d_out, int out_size, void* d_ws, size_t ws_size,
                              hipStream_t stream)
{
  const float* x  = (const float*)d_in[0];
  const float* Wq = (const float*)d_in[1];
  const float* sq = (const float*)d_in[2];
  const float* bq = (const float*)d_in[3];
  const float* Wk = (const float*)d_in[4];
  const float* sk = (const float*)d_in[5];
  const float* bk = (const float*)d_in[6];
  const float* Wv = (const float*)d_in[7];
  const float* sv = (const float*)d_in[8];
  const float* bv = (const float*)d_in[9];
  const float* Wf = (const float*)d_in[10];
  const float* bfv= (const float*)d_in[11];
  const float* sf = (const float*)d_in[12];
  const float* bff= (const float*)d_in[13];
  float* out = (float*)d_out;

  __bf16* xT  = (__bf16*)d_ws;                     // (B,N,C)   bf16
  __bf16* qT  = xT  + (size_t)BB*NN*CC;            // (B,N,C)   bf16
  __bf16* kT  = qT  + (size_t)BB*NN*CC;            // (B,N,C)   bf16
  __bf16* vN  = kT  + (size_t)BB*NN*CC;            // (B,C,N)   bf16
  __bf16* OT  = vN  + (size_t)BB*NN*CC;            // (B,N,448) bf16
  __bf16* Wqb = OT  + (size_t)BB*NN*FF;            // (256,256) bf16
  __bf16* Wkb = Wqb + (size_t)CC*CC;
  __bf16* Wvb = Wkb + (size_t)CC*CC;
  __bf16* Wfb = Wvb + (size_t)CC*CC;               // (256,448) bf16

  k_cvt<<<dim3((CC*CC+255)/256), 256, 0, stream>>>(Wq, Wqb, CC*CC);
  k_cvt<<<dim3((CC*CC+255)/256), 256, 0, stream>>>(Wk, Wkb, CC*CC);
  k_cvt<<<dim3((CC*CC+255)/256), 256, 0, stream>>>(Wv, Wvb, CC*CC);
  k_cvt<<<dim3((CC*FF+255)/256), 256, 0, stream>>>(Wf, Wfb, CC*FF);
  k_transpose<<<dim3(NN/32, CC/32, BB), dim3(32,8), 0, stream>>>(x, xT);
  k_qkv   <<<dim3(NN/128, CC/64, BB*3), 256, 0, stream>>>(xT, Wqb, Wkb, Wvb,
                                                          sq,bq, sk,bk, sv,bv,
                                                          qT, kT, vN);
  k_attn  <<<dim3(NN/32, HH, BB), 256, 0, stream>>>(qT, kT, vN, OT);
  k_final <<<dim3(NN/128, CC/64, BB), 256, 0, stream>>>(OT, Wfb, bfv, sf, bff, out);
}
